// GraphAttentionLayer_84610855731501
// MI455X (gfx1250) — compile-verified, matrix operand order checked by reading, not run on hardware
//
#include <hip/hip_runtime.h>
#include <hip/hip_bf16.h>
#include <math.h>

#define N_PTS 8192
#define CIN   256
#define COUT  512
#define NS    16
#define CW    64
#define NROWS (N_PTS * NS)
#define EPS_BN 1e-5f
#define PI_F 3.14159265358979323846f

typedef __bf16 v16bf __attribute__((ext_vector_type(16)));
typedef float  v8f   __attribute__((ext_vector_type(8)));

union Frag16 {
    v16bf v;
    unsigned short u[16];
    uint4 q[2];
};

__device__ __forceinline__ unsigned short f2bf(float f) {
    unsigned int u = __float_as_uint(f);
    unsigned int r = u + 0x7FFFu + ((u >> 16) & 1u);
    return (unsigned short)(r >> 16);
}

// low 32 bits of a generic pointer to LDS == workgroup-relative LDS offset
__device__ __forceinline__ unsigned lds_off(const void* p) {
    return (unsigned)(unsigned long long)p;
}

// ---------------------------------------------------------------------------
// Small prep kernels
// ---------------------------------------------------------------------------
__global__ void phys_kernel(const float* __restrict__ p, float* __restrict__ phys) {
    int i = blockIdx.x * blockDim.x + threadIdx.x;
    if (i >= N_PTS) return;
    float x = p[i * 3 + 0], y = p[i * 3 + 1];
    phys[i * 3 + 0] = sqrtf(x * x + y * y);
    phys[i * 3 + 1] = atan2f(y, x);
    phys[i * 3 + 2] = p[i * 3 + 2];
}

__global__ void cvt_bf16_kernel(const float* __restrict__ src,
                                unsigned short* __restrict__ dst, int n) {
    int i = blockIdx.x * blockDim.x + threadIdx.x;
    if (i < n) dst[i] = f2bf(src[i]);
}

// convert + transpose: src is K x Nn (row-major), dst is Nn x K (bf16).
// Transposed weights make WMMA B-fragments two contiguous b128 loads.
__global__ void cvt_bf16_t_kernel(const float* __restrict__ src,
                                  unsigned short* __restrict__ dst, int K, int Nn) {
    int id = blockIdx.x * blockDim.x + threadIdx.x;
    if (id >= K * Nn) return;
    int k = id / Nn, n0 = id - k * Nn;
    dst[(size_t)n0 * K + k] = f2bf(src[id]);
}

// ---------------------------------------------------------------------------
// KNN: one thread per query point. Candidate tiles staged into LDS with the
// CDNA5 async global->LDS DMA (ASYNCcnt), double-buffered so the next tile
// streams in while the current one is scanned. Register top-16 insertion.
// ---------------------------------------------------------------------------
__global__ void knn_kernel(const float* __restrict__ p,
                           int* __restrict__ idxb, float* __restrict__ distb) {
    int tid = threadIdx.x;
    int i = blockIdx.x * 256 + tid;
    float px = p[i * 3 + 0], py = p[i * 3 + 1], pz = p[i * 3 + 2];
    float bd[NS]; int bi[NS];
#pragma unroll
    for (int t = 0; t < NS; ++t) { bd[t] = 3.4e38f; bi[t] = 0; }

    __shared__ __align__(16) float tp[2][768];   // 256 points * xyz, 2 buffers

    // prefetch tile 0 (3072B = 192 x b128; threads 0..191 issue one each)
    if (tid < 192) {
        unsigned l = lds_off(&tp[0][tid * 4]);
        unsigned long long g = (unsigned long long)(p + tid * 4);
        asm volatile("global_load_async_to_lds_b128 %0, %1, off"
                     :: "v"(l), "v"(g) : "memory");
    }
    for (int tile = 0; tile < N_PTS / 256; ++tile) {
        int buf = tile & 1;
        asm volatile("s_wait_asynccnt 0" ::: "memory");
        __syncthreads();                    // this tile resident; prev scan done
        if (tile + 1 < N_PTS / 256 && tid < 192) {
            unsigned l = lds_off(&tp[buf ^ 1][tid * 4]);
            unsigned long long g =
                (unsigned long long)(p + (size_t)(tile + 1) * 768 + tid * 4);
            asm volatile("global_load_async_to_lds_b128 %0, %1, off"
                         :: "v"(l), "v"(g) : "memory");
        }
        const float* t = tp[buf];
        for (int jj = 0; jj < 256; ++jj) {
            float dx = px - t[jj * 3 + 0];
            float dy = py - t[jj * 3 + 1];
            float dz = pz - t[jj * 3 + 2];
            float d2 = dx * dx + dy * dy + dz * dz;
            if (d2 < bd[NS - 1]) {
                bd[NS - 1] = d2; bi[NS - 1] = tile * 256 + jj;
#pragma unroll
                for (int t2 = NS - 1; t2 > 0; --t2) {
                    if (bd[t2] < bd[t2 - 1]) {
                        float td = bd[t2]; bd[t2] = bd[t2 - 1]; bd[t2 - 1] = td;
                        int ti = bi[t2]; bi[t2] = bi[t2 - 1]; bi[t2 - 1] = ti;
                    }
                }
            }
        }
    }
#pragma unroll
    for (int t = 0; t < NS; ++t) {
        idxb[i * NS + t]  = bi[t];
        distb[i * NS + t] = sqrtf(bd[t] + 1e-12f);
    }
}

// ---------------------------------------------------------------------------
// Fused 3x GEMM: XQ = x@Wq + bq, XK = x@Wk, XV = x@Wv  (bf16 WMMA, f32 acc)
// Weights pre-transposed to [COUT][CIN]: A and B fragments are each two
// contiguous b128 loads (ISA 7.12.2 16-bit 16x32 layout: lane<16 holds
// K={kb..kb+7, 16+kb..16+kb+7}, kb = 0 / 8 by lane half).
// ---------------------------------------------------------------------------
__global__ __launch_bounds__(32)
void gemm3_kernel(const unsigned short* __restrict__ xb,
                  const unsigned short* __restrict__ wqt,
                  const unsigned short* __restrict__ wkt,
                  const unsigned short* __restrict__ wvt,
                  const float* __restrict__ bq,
                  float* __restrict__ XQ, float* __restrict__ XK,
                  float* __restrict__ XV) {
    int lane = threadIdx.x;
    int n0 = blockIdx.x * 16;
    int m0 = blockIdx.y * 16;
    int arow = m0 + (lane & 15);
    int ncol = n0 + (lane & 15);
    int kb = (lane < 16) ? 0 : 8;

    const unsigned short* arp = xb + (size_t)arow * CIN;
    const unsigned short* bqp = wqt + (size_t)ncol * CIN;
    const unsigned short* bkp = wkt + (size_t)ncol * CIN;
    const unsigned short* bvp = wvt + (size_t)ncol * CIN;

    v8f zero = {0.f, 0.f, 0.f, 0.f, 0.f, 0.f, 0.f, 0.f};
    v8f accq = zero, acck = zero, accv = zero;

    for (int kk = 0; kk < CIN / 32; ++kk) {
        int K0 = kk * 32;
        Frag16 A, Bq, Bk, Bv;
        A.q[0]  = *(const uint4*)(arp + K0 + kb);
        A.q[1]  = *(const uint4*)(arp + K0 + 16 + kb);
        Bq.q[0] = *(const uint4*)(bqp + K0 + kb);
        Bq.q[1] = *(const uint4*)(bqp + K0 + 16 + kb);
        Bk.q[0] = *(const uint4*)(bkp + K0 + kb);
        Bk.q[1] = *(const uint4*)(bkp + K0 + 16 + kb);
        Bv.q[0] = *(const uint4*)(bvp + K0 + kb);
        Bv.q[1] = *(const uint4*)(bvp + K0 + 16 + kb);
        accq = __builtin_amdgcn_wmma_f32_16x16x32_bf16(false, A.v, false, Bq.v,
                                                       (short)0, accq, false, false);
        acck = __builtin_amdgcn_wmma_f32_16x16x32_bf16(false, A.v, false, Bk.v,
                                                       (short)0, acck, false, false);
        accv = __builtin_amdgcn_wmma_f32_16x16x32_bf16(false, A.v, false, Bv.v,
                                                       (short)0, accv, false, false);
    }
    float biasq = bq[ncol];
#pragma unroll
    for (int v = 0; v < 8; ++v) {
        int m = m0 + v + ((lane < 16) ? 0 : 8);
        XQ[(size_t)m * COUT + ncol] = accq[v] + biasq;
        XK[(size_t)m * COUT + ncol] = acck[v];
        XV[(size_t)m * COUT + ncol] = accv[v];
    }
}

// ---------------------------------------------------------------------------
// Edge features -> pre-BN hidden (N*NS x 16) + BN1 stats accumulation
// ---------------------------------------------------------------------------
__global__ void edge_kernel(const float* __restrict__ physb,
                            const float* __restrict__ nrm,
                            const int* __restrict__ idxb,
                            const float* __restrict__ We1,
                            const float* __restrict__ be1,
                            float* __restrict__ hpre,
                            float* __restrict__ gsum, float* __restrict__ gss) {
    __shared__ float ssum[16], sss[16];
    int tid = threadIdx.x;
    if (tid < 16) { ssum[tid] = 0.f; sss[tid] = 0.f; }
    __syncthreads();
    int row = blockIdx.x * blockDim.x + tid;
    int i = row >> 4;
    int g = idxb[row];
    float pr = physb[g * 3 + 0] - physb[i * 3 + 0];
    float pt = physb[g * 3 + 1] - physb[i * 3 + 1];
    float pz = physb[g * 3 + 2] - physb[i * 3 + 2];
    float dr = fabsf(pr);
    float m = fmodf(pt + PI_F, 2.f * PI_F);
    if (m < 0.f) m += 2.f * PI_F;                 // jnp.mod semantics
    float dth = fabsf(m - PI_F);
    float dz = fabsf(pz);
    float nx = nrm[g * 3 + 0] - nrm[i * 3 + 0];
    float ny = nrm[g * 3 + 1] - nrm[i * 3 + 1];
    float nz = nrm[g * 3 + 2] - nrm[i * 3 + 2];
    float dn = sqrtf(nx * nx + ny * ny + nz * nz + 1e-12f);
#pragma unroll
    for (int k = 0; k < 16; ++k) {
        float h = be1[k] + dn * We1[0 * 16 + k] + dth * We1[1 * 16 + k]
                         + dz * We1[2 * 16 + k] + dr * We1[3 * 16 + k];
        hpre[(size_t)row * 16 + k] = h;
        atomicAdd(&ssum[k], h);
        atomicAdd(&sss[k], h * h);
    }
    __syncthreads();
    if (tid < 16) {
        atomicAdd(&gsum[tid], ssum[tid]);
        atomicAdd(&gss[tid], sss[tid]);
    }
}

// BN (train mode, biased var) folded to per-channel a*x + b
__global__ void bn_finalize_kernel(const float* __restrict__ sum,
                                   const float* __restrict__ ss,
                                   const float* __restrict__ gamma,
                                   const float* __restrict__ beta,
                                   float* __restrict__ a, float* __restrict__ b,
                                   int C, float inv_count) {
    int c = blockIdx.x * blockDim.x + threadIdx.x;
    if (c >= C) return;
    float mu = sum[c] * inv_count;
    float var = ss[c] * inv_count - mu * mu;
    float s = gamma[c] * rsqrtf(var + EPS_BN);
    a[c] = s;
    b[c] = beta[c] - mu * s;
}

// ---------------------------------------------------------------------------
// Pass 1 over w = x_q - x_k + emb : per-channel sum/sumsq (512 ch).
// One wave handles 16 points; lane owns channels {lane, 32+lane, ...}.
// ---------------------------------------------------------------------------
__global__ __launch_bounds__(32)
void wstat_kernel(const float* __restrict__ XQ, const float* __restrict__ XK,
                  const float* __restrict__ hpre, const int* __restrict__ idxb,
                  const float* __restrict__ We2, const float* __restrict__ be2,
                  const float* __restrict__ bk,
                  const float* __restrict__ a1e, const float* __restrict__ b1e,
                  float* __restrict__ wsum, float* __restrict__ wss) {
    int lane = threadIdx.x;
    __shared__ float hs[16][16];
    float sum[16], ss[16];
#pragma unroll
    for (int cc = 0; cc < 16; ++cc) { sum[cc] = 0.f; ss[cc] = 0.f; }
    int i0 = blockIdx.x * 16;
    for (int i = i0; i < i0 + 16; ++i) {
#pragma unroll
        for (int t = 0; t < 8; ++t) {
            int lin = lane * 8 + t;
            int r = lin >> 4, k = lin & 15;
            hs[r][k] = fmaxf(a1e[k] * hpre[(size_t)(i * NS + r) * 16 + k] + b1e[k], 0.f);
        }
        __syncthreads();
        for (int r = 0; r < NS; ++r) {
            int g = idxb[i * NS + r];
#pragma unroll
            for (int cc = 0; cc < 16; ++cc) {
                int c = cc * 32 + lane;
                float emb = be2[c];
#pragma unroll
                for (int k = 0; k < 16; ++k) emb += hs[r][k] * We2[k * COUT + c];
                float w = XQ[(size_t)i * COUT + c] - XK[(size_t)g * COUT + c]
                        + XK[(size_t)i * COUT + c] - bk[c] + emb;
                sum[cc] += w;
                ss[cc] += w * w;
            }
        }
        __syncthreads();
    }
#pragma unroll
    for (int cc = 0; cc < 16; ++cc) {
        atomicAdd(&wsum[cc * 32 + lane], sum[cc]);
        atomicAdd(&wss[cc * 32 + lane], ss[cc]);
    }
}

// ---------------------------------------------------------------------------
// Pass 2: recompute w, BN1+relu into LDS (bf16), then WMMA (16x512)x(512x64).
// Wc pre-transposed to [CW][COUT] so B fragments are two b128 loads.
// ---------------------------------------------------------------------------
__global__ __launch_bounds__(32)
void w64_kernel(const float* __restrict__ XQ, const float* __restrict__ XK,
                const float* __restrict__ hpre, const int* __restrict__ idxb,
                const float* __restrict__ We2, const float* __restrict__ be2,
                const float* __restrict__ bk,
                const float* __restrict__ a1e, const float* __restrict__ b1e,
                const float* __restrict__ a1, const float* __restrict__ b1,
                const unsigned short* __restrict__ wct,
                const float* __restrict__ bc, float* __restrict__ w64) {
    int lane = threadIdx.x;
    int i = blockIdx.x;
    __shared__ float hs[16][16];
    __shared__ __align__(16) unsigned short Als[16][COUT];
#pragma unroll
    for (int t = 0; t < 8; ++t) {
        int lin = lane * 8 + t;
        int r = lin >> 4, k = lin & 15;
        hs[r][k] = fmaxf(a1e[k] * hpre[(size_t)(i * NS + r) * 16 + k] + b1e[k], 0.f);
    }
    __syncthreads();
    for (int r = 0; r < NS; ++r) {
        int g = idxb[i * NS + r];
#pragma unroll
        for (int cc = 0; cc < 16; ++cc) {
            int c = cc * 32 + lane;
            float emb = be2[c];
#pragma unroll
            for (int k = 0; k < 16; ++k) emb += hs[r][k] * We2[k * COUT + c];
            float w = XQ[(size_t)i * COUT + c] - XK[(size_t)g * COUT + c]
                    + XK[(size_t)i * COUT + c] - bk[c] + emb;
            Als[r][c] = f2bf(fmaxf(a1[c] * w + b1[c], 0.f));
        }
    }
    __syncthreads();

    int arow = lane & 15;
    int kb = (lane < 16) ? 0 : 8;
    v8f zero = {0.f, 0.f, 0.f, 0.f, 0.f, 0.f, 0.f, 0.f};
    v8f acc[4] = {zero, zero, zero, zero};
    for (int kk = 0; kk < COUT / 32; ++kk) {
        int K0 = kk * 32;
        Frag16 A;
        A.q[0] = *(const uint4*)(&Als[arow][K0 + kb]);
        A.q[1] = *(const uint4*)(&Als[arow][K0 + 16 + kb]);
#pragma unroll
        for (int nt = 0; nt < 4; ++nt) {
            int ncol = nt * 16 + (lane & 15);
            const unsigned short* bp = wct + (size_t)ncol * COUT;
            Frag16 B;
            B.q[0] = *(const uint4*)(bp + K0 + kb);
            B.q[1] = *(const uint4*)(bp + K0 + 16 + kb);
            acc[nt] = __builtin_amdgcn_wmma_f32_16x16x32_bf16(false, A.v, false, B.v,
                                                              (short)0, acc[nt], false, false);
        }
    }
#pragma unroll
    for (int nt = 0; nt < 4; ++nt) {
        int ncol = nt * 16 + (lane & 15);
        float bias = bc[ncol];
#pragma unroll
        for (int v = 0; v < 8; ++v) {
            int m = v + ((lane < 16) ? 0 : 8);
            w64[(size_t)(i * NS + m) * CW + ncol] = acc[nt][v] + bias;
        }
    }
}

// BN2 stats over w64 (NROWS x 64)
__global__ void w64stat_kernel(const float* __restrict__ w64,
                               float* __restrict__ sum, float* __restrict__ ss) {
    __shared__ float lsum[CW], lss[CW];
    int tid = threadIdx.x;
    if (tid < CW) { lsum[tid] = 0.f; lss[tid] = 0.f; }
    __syncthreads();
    int c = tid & (CW - 1);
    float s = 0.f, q = 0.f;
    for (int r = blockIdx.x * 4 + (tid >> 6); r < NROWS; r += gridDim.x * 4) {
        float v = w64[(size_t)r * CW + c];
        s += v;
        q += v * v;
    }
    atomicAdd(&lsum[c], s);
    atomicAdd(&lss[c], q);
    __syncthreads();
    if (tid < CW) {
        atomicAdd(&sum[tid], lsum[tid]);
        atomicAdd(&ss[tid], lss[tid]);
    }
}

// BN2 + relu + (row x Ww) in place (each block owns one row)
__global__ __launch_bounds__(64)
void wfinal_kernel(float* __restrict__ w64, const float* __restrict__ a2,
                   const float* __restrict__ b2, const float* __restrict__ Ww,
                   const float* __restrict__ bw) {
    int row = blockIdx.x;
    int t = threadIdx.x;
    __shared__ float wn[CW];
    wn[t] = fmaxf(a2[t] * w64[(size_t)row * CW + t] + b2[t], 0.f);
    __syncthreads();
    float acc = bw[t];
#pragma unroll 8
    for (int k = 0; k < CW; ++k) acc += wn[k] * Ww[k * CW + t];
    w64[(size_t)row * CW + t] = acc;
}

// softmax over the NS neighbors for each (point, channel)
__global__ void softmax_kernel(float* __restrict__ w64) {
    int id = blockIdx.x * blockDim.x + threadIdx.x;
    if (id >= N_PTS * CW) return;
    int i = id >> 6, c = id & (CW - 1);
    float* base = w64 + (size_t)(i * NS) * CW + c;
    float mx = -3.4e38f;
#pragma unroll
    for (int j = 0; j < NS; ++j) mx = fmaxf(mx, base[j * CW]);
    float e[NS], s = 0.f;
#pragma unroll
    for (int j = 0; j < NS; ++j) { e[j] = expf(base[j * CW] - mx); s += e[j]; }
    float inv = 1.f / s;
#pragma unroll
    for (int j = 0; j < NS; ++j) base[j * CW] = e[j] * inv;
}

// final: out[i,c] = sum_j ((XV[g]-XV[i]+bv)*exp(-dist) + emb) * w[i,j,c%64]
__global__ __launch_bounds__(256)
void out_kernel(const float* __restrict__ XV, const float* __restrict__ hpre,
                const int* __restrict__ idxb, const float* __restrict__ distb,
                const float* __restrict__ We2, const float* __restrict__ be2,
                const float* __restrict__ bv,
                const float* __restrict__ a1e, const float* __restrict__ b1e,
                const float* __restrict__ wsm, float* __restrict__ out) {
    int i = blockIdx.x;
    int tid = threadIdx.x;
    __shared__ float hs[16][16];
    __shared__ float ed[NS];
    __shared__ int gi[NS];
    {
        int r = tid >> 4, k = tid & 15;
        hs[r][k] = fmaxf(a1e[k] * hpre[(size_t)(i * NS + r) * 16 + k] + b1e[k], 0.f);
    }
    if (tid < NS) {
        ed[tid] = expf(-distb[i * NS + tid]);
        gi[tid] = idxb[i * NS + tid];
    }
    __syncthreads();
#pragma unroll
    for (int half = 0; half < 2; ++half) {
        int c = tid + half * 256;
        float xvi = XV[(size_t)i * COUT + c];
        float bvc = bv[c];
        float acc = 0.f;
        for (int j = 0; j < NS; ++j) {
            float emb = be2[c];
#pragma unroll
            for (int k = 0; k < 16; ++k) emb += hs[j][k] * We2[k * COUT + c];
            float xv = (XV[(size_t)gi[j] * COUT + c] - xvi + bvc) * ed[j] + emb;
            acc += xv * wsm[(size_t)(i * NS + j) * CW + (c & (CW - 1))];
        }
        out[(size_t)i * COUT + c] = acc;
    }
}

// ---------------------------------------------------------------------------
extern "C" void kernel_launch(void* const* d_in, const int* in_sizes, int n_in,
                              void* d_out, int out_size, void* d_ws, size_t ws_size,
                              hipStream_t stream) {
    const float* p   = (const float*)d_in[0];
    const float* nrm = (const float*)d_in[1];
    const float* x   = (const float*)d_in[2];
    const float* Wq  = (const float*)d_in[3];
    const float* bq  = (const float*)d_in[4];
    const float* Wk  = (const float*)d_in[5];
    const float* bk  = (const float*)d_in[6];
    const float* Wv  = (const float*)d_in[7];
    const float* bv  = (const float*)d_in[8];
    const float* We1 = (const float*)d_in[9];
    const float* be1 = (const float*)d_in[10];
    const float* ge  = (const float*)d_in[11];
    const float* bbe = (const float*)d_in[12];
    const float* We2 = (const float*)d_in[13];
    const float* be2 = (const float*)d_in[14];
    const float* gw1 = (const float*)d_in[15];
    const float* bw1 = (const float*)d_in[16];
    const float* Wc  = (const float*)d_in[17];
    const float* bc  = (const float*)d_in[18];
    const float* gw2 = (const float*)d_in[19];
    const float* bw2 = (const float*)d_in[20];
    const float* Ww  = (const float*)d_in[21];
    const float* bw  = (const float*)d_in[22];

    char* base = (char*)d_ws;
    size_t off = 0;
    auto take = [&](size_t bytes) -> char* {
        char* r = base + off;
        off += (bytes + 255) & ~(size_t)255;
        return r;
    };
    float* physb = (float*)take((size_t)N_PTS * 3 * 4);
    int*   idxb  = (int*)take((size_t)NROWS * 4);
    float* distb = (float*)take((size_t)NROWS * 4);
    unsigned short* xbf  = (unsigned short*)take((size_t)N_PTS * CIN * 2);
    unsigned short* wqt  = (unsigned short*)take((size_t)CIN * COUT * 2);
    unsigned short* wkt  = (unsigned short*)take((size_t)CIN * COUT * 2);
    unsigned short* wvt  = (unsigned short*)take((size_t)CIN * COUT * 2);
    unsigned short* wct  = (unsigned short*)take((size_t)COUT * CW * 2);
    float* XQ = (float*)take((size_t)N_PTS * COUT * 4);
    float* XK = (float*)take((size_t)N_PTS * COUT * 4);
    float* XV = (float*)take((size_t)N_PTS * COUT * 4);
    float* hpre = (float*)take((size_t)NROWS * 16 * 4);
    float* stats = (float*)take(2368 * 4);
    float* w64 = (float*)take((size_t)NROWS * CW * 4);

    float* esum = stats;            // 16
    float* ess  = stats + 16;       // 16
    float* w1s  = stats + 32;       // 512
    float* w1ss = stats + 544;      // 512
    float* w2s  = stats + 1056;     // 64
    float* w2ss = stats + 1120;     // 64
    float* a1e  = stats + 1184;     // 16
    float* b1e  = stats + 1200;     // 16
    float* a1   = stats + 1216;     // 512
    float* b1   = stats + 1728;     // 512
    float* a2   = stats + 2240;     // 64
    float* b2   = stats + 2304;     // 64

    hipMemsetAsync(stats, 0, 1184 * sizeof(float), stream);

    phys_kernel<<<N_PTS / 256, 256, 0, stream>>>(p, physb);
    cvt_bf16_kernel<<<(N_PTS * CIN + 255) / 256, 256, 0, stream>>>(x, xbf, N_PTS * CIN);
    cvt_bf16_t_kernel<<<(CIN * COUT + 255) / 256, 256, 0, stream>>>(Wq, wqt, CIN, COUT);
    cvt_bf16_t_kernel<<<(CIN * COUT + 255) / 256, 256, 0, stream>>>(Wk, wkt, CIN, COUT);
    cvt_bf16_t_kernel<<<(CIN * COUT + 255) / 256, 256, 0, stream>>>(Wv, wvt, CIN, COUT);
    cvt_bf16_t_kernel<<<(COUT * CW + 255) / 256, 256, 0, stream>>>(Wc, wct, COUT, CW);

    knn_kernel<<<N_PTS / 256, 256, 0, stream>>>(p, idxb, distb);

    gemm3_kernel<<<dim3(COUT / 16, N_PTS / 16), 32, 0, stream>>>(
        xbf, wqt, wkt, wvt, bq, XQ, XK, XV);

    edge_kernel<<<NROWS / 256, 256, 0, stream>>>(physb, nrm, idxb, We1, be1,
                                                 hpre, esum, ess);
    bn_finalize_kernel<<<1, 16, 0, stream>>>(esum, ess, ge, bbe, a1e, b1e, 16,
                                             1.f / (float)NROWS);

    wstat_kernel<<<N_PTS / 16, 32, 0, stream>>>(XQ, XK, hpre, idxb, We2, be2, bk,
                                                a1e, b1e, w1s, w1ss);
    bn_finalize_kernel<<<2, 256, 0, stream>>>(w1s, w1ss, gw1, bw1, a1, b1, COUT,
                                              1.f / (float)NROWS);

    w64_kernel<<<N_PTS, 32, 0, stream>>>(XQ, XK, hpre, idxb, We2, be2, bk,
                                         a1e, b1e, a1, b1, wct, bc, w64);

    w64stat_kernel<<<512, 256, 0, stream>>>(w64, w2s, w2ss);
    bn_finalize_kernel<<<1, 64, 0, stream>>>(w2s, w2ss, gw2, bw2, a2, b2, CW,
                                             1.f / (float)NROWS);

    wfinal_kernel<<<NROWS, 64, 0, stream>>>(w64, a2, b2, Ww, bw);
    softmax_kernel<<<(N_PTS * CW + 255) / 256, 256, 0, stream>>>(w64);

    out_kernel<<<N_PTS, 256, 0, stream>>>(XV, hpre, idxb, distb, We2, be2, bv,
                                          a1e, b1e, w64, (float*)d_out);
}
